// AttentionBlock_80685255622698
// MI455X (gfx1250) — matrix operand order repaired and checked by reading, not compiled, not run on hardware
//
#include <hip/hip_runtime.h>
#include <hip/hip_fp16.h>

typedef __attribute__((ext_vector_type(16))) _Float16 v16h;
typedef __attribute__((ext_vector_type(8)))  _Float16 v8h;
typedef __attribute__((ext_vector_type(8)))  float    v8f;
typedef __attribute__((ext_vector_type(4)))  unsigned int uint32x4;
typedef __attribute__((ext_vector_type(8)))  int          int32x8;
typedef __attribute__((ext_vector_type(4)))  int          int32x4;

#define BB   8
#define CC   256
#define TT   4096
#define CQK  32
#define OTOT 320     // 32 (q) + 32 (k) + 256 (v)

// ---------------------------------------------------------------------------
// TDM: DMA a 64(s-rows) x 32(halves) = 4KB K tile from global into LDS.
// D# bit layout per CDNA5 ISA 8.3/8.4:
//   g0: [1:0]=count=1 | [63:32]=lds_addr | [120:64]=global_addr | [127:126]=type=2
//   g1: [17:16]=data_size=1 (2B) | [79:48]=tensor_dim0=32 | [111:80]=tensor_dim1=64
//       [127:112]=tile_dim0=32 | [143:128]=tile_dim1=64 | [207:160]=dim0_stride=32
//   g2/g3 (+trailing group): zero (2D tile)
// Toolchain uses the 6-arg builtin: (v4u, v8i, v4i, v4i, v8i, i32 cpol).
// ---------------------------------------------------------------------------
__device__ __forceinline__ void tdm_load_k_tile(const _Float16* gptr,
                                                unsigned lds_byte_off) {
    unsigned long long ga = (unsigned long long)(const void*)gptr;
    uint32x4 g0;
    g0[0] = 1u;                                        // count=1, user descriptor
    g0[1] = lds_byte_off;                              // lds_addr (bytes)
    g0[2] = (unsigned)(ga & 0xFFFFFFFFu);              // global_addr[31:0]
    g0[3] = (unsigned)((ga >> 32) & 0x01FFFFFFu)       // global_addr[56:32]
          | (2u << 30);                                // type = 2 ("image")
    int32x8 g1;
    g1[0] = (int)(1u << 16);                           // data_size = 1 -> 2 bytes
    g1[1] = (int)(32u << 16);                          // tensor_dim0[15:0] = 32
    g1[2] = (int)(64u << 16);                          // tensor_dim1[15:0] = 64
    g1[3] = (int)(32u << 16);                          // tile_dim0 = 32
    g1[4] = (int)64u;                                  // tile_dim1 = 64
    g1[5] = (int)32u;                                  // tensor_dim0_stride = 32
    g1[6] = 0;
    g1[7] = 0;
    int32x4 gz4 = {0, 0, 0, 0};
    int32x8 gz8 = {0, 0, 0, 0, 0, 0, 0, 0};
    __builtin_amdgcn_tensor_load_to_lds(g0, g1, gz4, gz4, gz8, 0);
}

// ---------------------------------------------------------------------------
// Kernel 1: fused q/k/v projection (one wave = 16(o) x 16(t) tile,
// 8 chained v_wmma_f32_16x16x32_f16 over C=256).
// q,k -> f16 [B][T][32] row-contiguous; v -> f16 [B][C][T] (s-contiguous).
// ---------------------------------------------------------------------------
__global__ __launch_bounds__(256) void proj_kernel(
    const float* __restrict__ x,
    const float* __restrict__ Wq, const float* __restrict__ bq,
    const float* __restrict__ Wk, const float* __restrict__ bk,
    const float* __restrict__ Wv, const float* __restrict__ bv,
    _Float16* __restrict__ qh, _Float16* __restrict__ kh,
    _Float16* __restrict__ vhT)
{
    const int lane = threadIdx.x & 31;
    const int wave = threadIdx.x >> 5;
    // wave-uniform tile id -> scalar branches, no EXEC ladders in epilogue
    const int tile = __builtin_amdgcn_readfirstlane(blockIdx.x * 8 + wave);

    const int ttile = tile & 255;                     // T/16
    const int otile = (tile >> 8) % (OTOT / 16);      // 20
    const int b     = tile / (256 * (OTOT / 16));

    const int half = lane >> 4;
    const int lid  = lane & 15;
    const int tcol = ttile * 16 + lid;

    const int o0 = otile * 16;
    const float* W;  const float* bias;  int obase, region;
    if (o0 < 32)       { W = Wq; bias = bq; obase = o0;      region = 0; }
    else if (o0 < 64)  { W = Wk; bias = bk; obase = o0 - 32; region = 1; }
    else               { W = Wv; bias = bv; obase = o0 - 64; region = 2; }

    const int arow  = obase + lid;
    const int aoff1 = half ? 16 : 0;
    const int aoff2 = half ? 24 : 8;
    const int koff  = half ? 16 : 0;

    const float* xb = x + (size_t)b * CC * TT;

    v8f acc = {};
    for (int c0 = 0; c0 < CC; c0 += 32) {
        v16h a, bm;
        const float* wr = W + (size_t)arow * CC + c0;
        #pragma unroll
        for (int j = 0; j < 8; ++j) {
            a[j]     = (_Float16)wr[aoff1 + j];
            a[j + 8] = (_Float16)wr[aoff2 + j];
        }
        #pragma unroll
        for (int j = 0; j < 16; ++j)
            bm[j] = (_Float16)xb[(size_t)(c0 + koff + j) * TT + tcol];
        acc = __builtin_amdgcn_wmma_f32_16x16x32_f16(
                  false, a, false, bm, (short)0, acc, false, false);
    }

    #pragma unroll
    for (int r = 0; r < 8; ++r) {
        const int orow = obase + r + half * 8;
        const float v  = acc[r] + bias[orow];
        if (region == 0)
            qh[((size_t)b * TT + tcol) * CQK + orow] = (_Float16)v;
        else if (region == 1)
            kh[((size_t)b * TT + tcol) * CQK + orow] = (_Float16)v;
        else
            vhT[((size_t)b * CC + orow) * TT + tcol] = (_Float16)v;
    }
}

// ---------------------------------------------------------------------------
// Kernel 2: fused flash attention + residual.
// 4 waves/block; one wave = 16 query rows x 128 channels, s-tile = 64.
// K tiles: TDM double-buffered into LDS (tensor_load_to_lds/s_wait_tensorcnt).
// Per iteration: 4 score WMMAs + online softmax + P transpose via LDS +
// 16 output WMMAs. Epilogue: gamma*O/l + x.
// LDS: 4 waves * (2*4KB K + 2KB P) = 40KB.
// ---------------------------------------------------------------------------
#define KBUF_BYTES   4096
#define KREG_BYTES   (2 * KBUF_BYTES)        // per-wave double buffer
#define PBASE_BYTES  (4 * KREG_BYTES)        // 32768
#define SMEM_BYTES   (PBASE_BYTES + 4 * 2048)

__global__ __launch_bounds__(128) void attn_kernel(
    const _Float16* __restrict__ qh, const _Float16* __restrict__ kh,
    const _Float16* __restrict__ vhT, const float* __restrict__ x,
    const float* __restrict__ gamma, float* __restrict__ out)
{
    __shared__ char smem[SMEM_BYTES];

    const int lane  = threadIdx.x & 31;
    const int wave  = threadIdx.x >> 5;
    const int id    = blockIdx.x * 4 + wave;          // 4096 waves total
    const int chalf = id & 1;
    const int qtile = (id >> 1) & 255;
    const int b     = id >> 9;

    const int half = lane >> 4;
    const int lid  = lane & 15;
    const int tq0  = qtile * 16;
    const int c0   = chalf * 128;

    const unsigned kbase = (unsigned)(wave * KREG_BYTES);
    _Float16* P = (_Float16*)(smem + PBASE_BYTES + wave * 2048);  // 16 x 64 f16

    // Q tile as A-operand, loaded once (rows contiguous in qh)
    v16h qa;
    {
        const _Float16* qrow = qh + ((size_t)b * TT + (tq0 + lid)) * CQK;
        const int o1 = half ? 16 : 0, o2 = half ? 24 : 8;
        v8h lo = *(const v8h*)(qrow + o1);
        v8h hi = *(const v8h*)(qrow + o2);
        #pragma unroll
        for (int j = 0; j < 8; ++j) { qa[j] = lo[j]; qa[j + 8] = hi[j]; }
    }

    v8f Oacc[8] = {};
    float m[8], l[8];
    #pragma unroll
    for (int r = 0; r < 8; ++r) { m[r] = -3.0e38f; l[r] = 0.0f; }

    const _Float16* kbat = kh + (size_t)b * TT * CQK;

    // prologue: DMA first K tile into buffer 0
    tdm_load_k_tile(kbat, kbase);

    for (int s0 = 0; s0 < TT; s0 += 64) {
        const int buf = (s0 >> 6) & 1;
        if (s0 + 64 < TT) {           // prefetch next tile into other buffer
            tdm_load_k_tile(kbat + (size_t)(s0 + 64) * CQK,
                            kbase + (unsigned)((buf ^ 1) * KBUF_BYTES));
            __builtin_amdgcn_s_wait_tensorcnt((short)1);
        } else {
            __builtin_amdgcn_s_wait_tensorcnt((short)0);
        }
        const _Float16* Kt = (const _Float16*)(smem + kbase + buf * KBUF_BYTES);

        // ---- scores: S[16 x 64] as four 16x16 f32 tiles (K from LDS) ------
        v8f S[4];
        #pragma unroll
        for (int nt = 0; nt < 4; ++nt) {
            const _Float16* krow = Kt + (nt * 16 + lid) * CQK + (half ? 16 : 0);
            v16h kb = *(const v16h*)krow;
            v8f z = {};
            S[nt] = __builtin_amdgcn_wmma_f32_16x16x32_f16(
                        false, qa, false, kb, (short)0, z, false, false);
        }

        // ---- online softmax (row stats on 16-lane half-rows) --------------
        #pragma unroll
        for (int r = 0; r < 8; ++r) {
            float mt = fmaxf(fmaxf(S[0][r], S[1][r]), fmaxf(S[2][r], S[3][r]));
            mt = fmaxf(mt, __shfl_xor(mt, 1, 32));
            mt = fmaxf(mt, __shfl_xor(mt, 2, 32));
            mt = fmaxf(mt, __shfl_xor(mt, 4, 32));
            mt = fmaxf(mt, __shfl_xor(mt, 8, 32));
            const float mnew = fmaxf(m[r], mt);
            const float sc   = __expf(m[r] - mnew);
            m[r] = mnew;

            S[0][r] = __expf(S[0][r] - mnew);
            S[1][r] = __expf(S[1][r] - mnew);
            S[2][r] = __expf(S[2][r] - mnew);
            S[3][r] = __expf(S[3][r] - mnew);
            float rs = (S[0][r] + S[1][r]) + (S[2][r] + S[3][r]);
            rs += __shfl_xor(rs, 1, 32);
            rs += __shfl_xor(rs, 2, 32);
            rs += __shfl_xor(rs, 4, 32);
            rs += __shfl_xor(rs, 8, 32);
            l[r] = l[r] * sc + rs;

            #pragma unroll
            for (int ct = 0; ct < 8; ++ct) Oacc[ct][r] *= sc;
        }

        // ---- P transpose (C-layout -> A-layout) via per-wave LDS ----------
        #pragma unroll
        for (int r = 0; r < 8; ++r) {
            const int row = r + half * 8;
            #pragma unroll
            for (int nt = 0; nt < 4; ++nt)
                P[row * 64 + nt * 16 + lid] = (_Float16)S[nt][r];
        }
        asm volatile("s_wait_dscnt 0" ::: "memory");   // wave-local LDS RAW

        v16h pa[2];
        #pragma unroll
        for (int sub = 0; sub < 2; ++sub) {
            const _Float16* prow = P + lid * 64 + sub * 32;
            const int o1 = half ? 16 : 0, o2 = half ? 24 : 8;
            v8h lo = *(const v8h*)(prow + o1);
            v8h hi = *(const v8h*)(prow + o2);
            #pragma unroll
            for (int j = 0; j < 8; ++j) { pa[sub][j] = lo[j]; pa[sub][j + 8] = hi[j]; }
        }

        // ---- O += P @ V  (8 channel tiles x 2 k-subtiles) -----------------
        #pragma unroll
        for (int ct = 0; ct < 8; ++ct) {
            const int c = c0 + ct * 16 + lid;
            const _Float16* vrow = vhT + ((size_t)b * CC + c) * TT + s0
                                   + (half ? 16 : 0);
            v16h vb0 = *(const v16h*)vrow;
            v16h vb1 = *(const v16h*)(vrow + 32);
            Oacc[ct] = __builtin_amdgcn_wmma_f32_16x16x32_f16(
                           false, pa[0], false, vb0, (short)0, Oacc[ct], false, false);
            Oacc[ct] = __builtin_amdgcn_wmma_f32_16x16x32_f16(
                           false, pa[1], false, vb1, (short)0, Oacc[ct], false, false);
        }

        // cache prefetch for next V stripe (global_prefetch_b8)
        if (s0 + 64 < TT)
            __builtin_prefetch(
                vhT + ((size_t)b * CC + (c0 + lid)) * TT + s0 + 64, 0, 1);
    }

    // ---- epilogue: out = gamma * (O / l) + x ------------------------------
    const float g = gamma[0];
    #pragma unroll
    for (int ct = 0; ct < 8; ++ct) {
        const int c = c0 + ct * 16 + lid;
        #pragma unroll
        for (int r = 0; r < 8; ++r) {
            const int t = tq0 + r + half * 8;
            const size_t idx = ((size_t)b * CC + c) * TT + t;
            out[idx] = g * (Oacc[ct][r] / l[r]) + x[idx];
        }
    }
}

// ---------------------------------------------------------------------------
extern "C" void kernel_launch(void* const* d_in, const int* in_sizes, int n_in,
                              void* d_out, int out_size, void* d_ws, size_t ws_size,
                              hipStream_t stream) {
    const float* x     = (const float*)d_in[0];
    const float* Wq    = (const float*)d_in[1];
    const float* bq    = (const float*)d_in[2];
    const float* Wk    = (const float*)d_in[3];
    const float* bk    = (const float*)d_in[4];
    const float* Wv    = (const float*)d_in[5];
    const float* bv    = (const float*)d_in[6];
    const float* gamma = (const float*)d_in[7];
    float* out = (float*)d_out;

    // workspace: qh 2MB | kh 2MB | vhT 16MB (all f16)
    char* ws = (char*)d_ws;
    _Float16* qh  = (_Float16*)(ws);
    _Float16* kh  = (_Float16*)(ws + (size_t)2 * 1024 * 1024);
    _Float16* vhT = (_Float16*)(ws + (size_t)4 * 1024 * 1024);

    proj_kernel<<<5120, 256, 0, stream>>>(x, Wq, bq, Wk, bk, Wv, bv,
                                          qh, kh, vhT);
    attn_kernel<<<1024, 128, 0, stream>>>(qh, kh, vhT, x, gamma, out);
}
